// NarrowSelfAttention_40845138985149
// MI455X (gfx1250) — matrix operand order repaired and checked by reading, not compiled, not run on hardware
//
#include <hip/hip_runtime.h>
#include <hip/hip_bf16.h>

// ---------------------------------------------------------------------------
// NarrowSelfAttention for MI455X (gfx1250): bf16 WMMA flash attention
// ---------------------------------------------------------------------------
#define B_      4
#define T_      2048
#define EMB_    1024
#define HEADS_  16
#define CHUNK_  64
#define SCALE_  0.176776695296637f   // EMB^-0.25

typedef __attribute__((ext_vector_type(16))) __bf16 v16bf;
typedef __attribute__((ext_vector_type(8)))  float  v8f;

static __device__ __forceinline__ v8f wmma_bf16(v16bf a, v16bf b, v8f c) {
  // D = A(16x32) * B(32x16) + C, f32 accumulate
  return __builtin_amdgcn_wmma_f32_16x16x32_bf16(
      /*neg_a=*/false, a, /*neg_b=*/false, b,
      /*c_mod=*/(short)0, c, /*reuse_a=*/false, /*reuse_b=*/false);
}

// A-operand fill used here (contiguous k; B loads match): lane holds row m = lane&15;
// VGPR i holds K pair: kbase + kh + 2i, kh = (lane>>4)*16.
static __device__ __forceinline__ v16bf load_a_bf16(const __bf16* row, int lane, int kbase) {
  const int kh = (lane >> 4) << 4;
  v16bf a;
#pragma unroll
  for (int i = 0; i < 8; ++i) {
    int k = kbase + kh + (i << 1);
    union { unsigned u; __bf16 h[2]; } cv;
    cv.u = *(const unsigned*)(row + k);   // 4B aligned (k even)
    a[2 * i]     = cv.h[0];
    a[2 * i + 1] = cv.h[1];
  }
  return a;
}

static __device__ __forceinline__ v16bf load_a_f32(const float* row, int lane, int kbase) {
  const int kh = (lane >> 4) << 4;
  v16bf a;
#pragma unroll
  for (int i = 0; i < 8; ++i) {
    int k = kbase + kh + (i << 1);
    a[2 * i]     = (__bf16)row[k];
    a[2 * i + 1] = (__bf16)row[k + 1];
  }
  return a;
}

// ---------------------------------------------------------------------------
// Stage 0: fp32 -> bf16 weight conversion
// ---------------------------------------------------------------------------
__global__ void cvt_kernel(const float* __restrict__ s, __bf16* __restrict__ d, int nElem) {
  int i = blockIdx.x * blockDim.x + threadIdx.x;
  if (i < nElem) d[i] = (__bf16)s[i];
}

// ---------------------------------------------------------------------------
// Stage 1: QKV projection.  One wave per 16-row tile of one (b,h).
// y = x_head @ W.T (W is (out,in)) -> B-frag[k=c][n=o] = W[o][c] (row-contig).
// Q,K scaled by EMB^-0.25; V stored transposed (C x T) per (b,h).
// ---------------------------------------------------------------------------
__global__ void __launch_bounds__(256)
qkv_kernel(const float* __restrict__ x,
           const __bf16* __restrict__ Wqb, const __bf16* __restrict__ Wkb,
           const __bf16* __restrict__ Wvb,
           __bf16* __restrict__ Qb, __bf16* __restrict__ Kb,
           __bf16* __restrict__ Vt) {
  const int lane  = threadIdx.x & 31;
  const int wave  = threadIdx.x >> 5;
  const int gtile = blockIdx.x * 8 + wave;           // 8192 tiles total
  const int ntt   = T_ / 16;
  const int tt = gtile % ntt;
  const int bh = gtile / ntt;                        // b*HEADS_+h
  const int h  = bh % HEADS_;
  const int b  = bh / HEADS_;
  const int t0 = tt * 16;
  const int n     = lane & 15;
  const int kb    = (lane >> 4) << 4;
  const int mbase = (lane >> 4) << 3;
  const int m     = lane & 15;

  const float* xrow = x + ((size_t)(b * T_ + t0 + m) * EMB_ + h * CHUNK_);
  v16bf a0 = load_a_f32(xrow, lane, 0);
  v16bf a1 = load_a_f32(xrow, lane, 32);

  const size_t bhT = (size_t)bh * T_;
#pragma unroll
  for (int wi = 0; wi < 3; ++wi) {
    const __bf16* Wb = (wi == 0) ? Wqb : (wi == 1) ? Wkb : Wvb;
#pragma unroll
    for (int cc = 0; cc < 4; ++cc) {
      v8f acc = {};
      const __bf16* wrow = Wb + (cc * 16 + n) * CHUNK_ + kb;  // 32B aligned
      acc = wmma_bf16(a0, *(const v16bf*)(wrow), acc);
      acc = wmma_bf16(a1, *(const v16bf*)(wrow + 32), acc);
#pragma unroll
      for (int v = 0; v < 8; ++v) {
        const int row = t0 + mbase + v;
        const int col = cc * 16 + n;
        if (wi == 0)      Qb[(bhT + row) * CHUNK_ + col] = (__bf16)(acc[v] * SCALE_);
        else if (wi == 1) Kb[(bhT + row) * CHUNK_ + col] = (__bf16)(acc[v] * SCALE_);
        else              Vt[((size_t)bh * CHUNK_ + col) * T_ + row] = (__bf16)acc[v];
      }
    }
  }
}

// ---------------------------------------------------------------------------
// Stage 2: causal flash attention.  One wave per 16-query tile; key blocks
// of 32 (two 16x16 S tiles).  Online softmax in f32; P goes through LDS to
// convert C-layout -> A-layout; O = P@V accumulated in four 16x16 tiles.
// ---------------------------------------------------------------------------
__global__ void __launch_bounds__(256)
attn_kernel(const __bf16* __restrict__ Qb, const __bf16* __restrict__ Kb,
            const __bf16* __restrict__ Vt, __bf16* __restrict__ Yb) {
  __shared__ __bf16 pbuf[8][16][32];                 // per-wave P staging (8 KB)
  const int lane  = threadIdx.x & 31;
  const int wave  = threadIdx.x >> 5;
  const int gtile = blockIdx.x * 8 + wave;
  const int ntt   = T_ / 16;
  const int tt = gtile % ntt;
  const int bh = gtile / ntt;
  const int q0 = tt * 16;
  const int n     = lane & 15;
  const int kb    = (lane >> 4) << 4;
  const int mbase = (lane >> 4) << 3;
  const int m     = lane & 15;
  const size_t bhT = (size_t)bh * T_;

  const __bf16* qrow = Qb + (bhT + q0 + m) * CHUNK_;
  v16bf aq0 = load_a_bf16(qrow, lane, 0);
  v16bf aq1 = load_a_bf16(qrow, lane, 32);

  v8f o0 = {}, o1 = {}, o2 = {}, o3 = {};
  float mrow[8], lrow[8];
#pragma unroll
  for (int v = 0; v < 8; ++v) { mrow[v] = -__builtin_inff(); lrow[v] = 0.f; }

  for (int jb = 0; jb < q0 + 16; jb += 32) {
    // ---- S = Q K^T for keys [jb, jb+32) : B-frag[k=c][n=key] = K[key][c]
    v8f s0 = {}, s1 = {};
    const __bf16* k0row = Kb + (bhT + jb + n) * CHUNK_ + kb;
    const __bf16* k1row = Kb + (bhT + jb + 16 + n) * CHUNK_ + kb;
    s0 = wmma_bf16(aq0, *(const v16bf*)(k0row), s0);
    s0 = wmma_bf16(aq1, *(const v16bf*)(k0row + 32), s0);
    s1 = wmma_bf16(aq0, *(const v16bf*)(k1row), s1);
    s1 = wmma_bf16(aq1, *(const v16bf*)(k1row + 32), s1);

    // ---- causal mask + online softmax (C-layout: lane=col, 8 rows/lane)
    float alpha[8];
#pragma unroll
    for (int v = 0; v < 8; ++v) {
      const int row = q0 + mbase + v;
      float x0 = (jb + n      > row) ? -__builtin_inff() : s0[v];
      float x1 = (jb + 16 + n > row) ? -__builtin_inff() : s1[v];
      float bm = fmaxf(x0, x1);
#pragma unroll
      for (int msk = 1; msk < 16; msk <<= 1) bm = fmaxf(bm, __shfl_xor(bm, msk, 32));
      const float mn = fmaxf(mrow[v], bm);
      const float p0 = __expf(x0 - mn);
      const float p1 = __expf(x1 - mn);
      float rs = p0 + p1;
#pragma unroll
      for (int msk = 1; msk < 16; msk <<= 1) rs += __shfl_xor(rs, msk, 32);
      alpha[v] = __expf(mrow[v] - mn);
      mrow[v]  = mn;
      lrow[v]  = lrow[v] * alpha[v] + rs;
      pbuf[wave][mbase + v][n]      = (__bf16)p0;    // C-layout -> LDS
      pbuf[wave][mbase + v][16 + n] = (__bf16)p1;
    }
#pragma unroll
    for (int v = 0; v < 8; ++v) {
      o0[v] *= alpha[v]; o1[v] *= alpha[v]; o2[v] *= alpha[v]; o3[v] *= alpha[v];
    }
    asm volatile("s_wait_dscnt 0" ::: "memory");     // all lanes' P visible
    v16bf pa = load_a_bf16(&pbuf[wave][m][0], lane, 0);  // LDS -> A-layout

    // ---- O += P @ V : B-frag[k=key][n=c] = Vt[c][key] (contiguous keys)
    const __bf16* vbase = Vt + ((size_t)bh * CHUNK_ + n) * T_ + jb + kb;
    o0 = wmma_bf16(pa, *(const v16bf*)(vbase),            o0);
    o1 = wmma_bf16(pa, *(const v16bf*)(vbase + 16 * T_),  o1);
    o2 = wmma_bf16(pa, *(const v16bf*)(vbase + 32 * T_),  o2);
    o3 = wmma_bf16(pa, *(const v16bf*)(vbase + 48 * T_),  o3);
  }

  // ---- normalize and store to (b, t, h*64 + c) bf16 for the final GEMM
  const int h = bh % HEADS_;
  const int b = bh / HEADS_;
#pragma unroll
  for (int v = 0; v < 8; ++v) {
    const float inv = 1.f / lrow[v];
    const size_t base = (size_t)(b * T_ + q0 + mbase + v) * EMB_ + h * CHUNK_;
    Yb[base + n]      = (__bf16)(o0[v] * inv);
    Yb[base + 16 + n] = (__bf16)(o1[v] * inv);
    Yb[base + 32 + n] = (__bf16)(o2[v] * inv);
    Yb[base + 48 + n] = (__bf16)(o3[v] * inv);
  }
}

// ---------------------------------------------------------------------------
// Stage 3: out = Y @ Wu.T + bu  (8192x1024 x 1024x1024), 16x64 tile per wave.
// ---------------------------------------------------------------------------
__global__ void __launch_bounds__(256)
proj_kernel(const __bf16* __restrict__ Yb, const __bf16* __restrict__ Wub,
            const float* __restrict__ bu, float* __restrict__ out) {
  const int lane  = threadIdx.x & 31;
  const int wave  = threadIdx.x >> 5;
  const int gtile = blockIdx.x * 8 + wave;           // 512 x 16 tiles
  const int ot = gtile & 15;
  const int mt = gtile >> 4;
  const int r0  = mt * 16;
  const int c0  = ot * 64;
  const int n     = lane & 15;
  const int kb    = (lane >> 4) << 4;
  const int mbase = (lane >> 4) << 3;
  const int m     = lane & 15;

  v8f acc[4] = {v8f{}, v8f{}, v8f{}, v8f{}};
  const __bf16* yrow = Yb + (size_t)(r0 + m) * EMB_;
  for (int kc = 0; kc < EMB_; kc += 32) {
    v16bf a = load_a_bf16(yrow, lane, kc);
#pragma unroll
    for (int cc = 0; cc < 4; ++cc) {
      const __bf16* wrow = Wub + (size_t)(c0 + cc * 16 + n) * EMB_ + kc + kb;
      acc[cc] = wmma_bf16(a, *(const v16bf*)wrow, acc[cc]);
    }
  }
#pragma unroll
  for (int cc = 0; cc < 4; ++cc) {
    const int col  = c0 + cc * 16 + n;
    const float bias = bu[col];
#pragma unroll
    for (int v = 0; v < 8; ++v)
      out[(size_t)(r0 + mbase + v) * EMB_ + col] = acc[cc][v] + bias;
  }
}

// ---------------------------------------------------------------------------
extern "C" void kernel_launch(void* const* d_in, const int* in_sizes, int n_in,
                              void* d_out, int out_size, void* d_ws, size_t ws_size,
                              hipStream_t stream) {
  (void)in_sizes; (void)n_in; (void)out_size; (void)ws_size;
  const float* x  = (const float*)d_in[0];
  const float* Wq = (const float*)d_in[1];
  const float* Wk = (const float*)d_in[2];
  const float* Wv = (const float*)d_in[3];
  const float* Wu = (const float*)d_in[4];
  const float* bu = (const float*)d_in[5];
  float* out = (float*)d_out;

  char* ws = (char*)d_ws;
  size_t off = 0;
  auto alloc = [&](size_t bytes) {
    void* p = ws + off;
    off = (off + bytes + 255) & ~(size_t)255;
    return p;
  };
  const size_t nQKV = (size_t)B_ * HEADS_ * T_ * CHUNK_;   // 8,388,608
  __bf16* Wqb = (__bf16*)alloc(CHUNK_ * CHUNK_ * sizeof(__bf16));
  __bf16* Wkb = (__bf16*)alloc(CHUNK_ * CHUNK_ * sizeof(__bf16));
  __bf16* Wvb = (__bf16*)alloc(CHUNK_ * CHUNK_ * sizeof(__bf16));
  __bf16* Wub = (__bf16*)alloc((size_t)EMB_ * EMB_ * sizeof(__bf16));
  __bf16* Qb  = (__bf16*)alloc(nQKV * sizeof(__bf16));
  __bf16* Kb  = (__bf16*)alloc(nQKV * sizeof(__bf16));
  __bf16* Vt  = (__bf16*)alloc(nQKV * sizeof(__bf16));
  __bf16* Yb  = (__bf16*)alloc((size_t)B_ * T_ * EMB_ * sizeof(__bf16));

  cvt_kernel<<<(CHUNK_ * CHUNK_ + 255) / 256, 256, 0, stream>>>(Wq, Wqb, CHUNK_ * CHUNK_);
  cvt_kernel<<<(CHUNK_ * CHUNK_ + 255) / 256, 256, 0, stream>>>(Wk, Wkb, CHUNK_ * CHUNK_);
  cvt_kernel<<<(CHUNK_ * CHUNK_ + 255) / 256, 256, 0, stream>>>(Wv, Wvb, CHUNK_ * CHUNK_);
  cvt_kernel<<<(EMB_ * EMB_ + 255) / 256, 256, 0, stream>>>(Wu, Wub, EMB_ * EMB_);

  const int tiles = B_ * HEADS_ * (T_ / 16);   // 8192 wave-tiles
  qkv_kernel<<<tiles / 8, 256, 0, stream>>>(x, Wqb, Wkb, Wvb, Qb, Kb, Vt);
  attn_kernel<<<tiles / 8, 256, 0, stream>>>(Qb, Kb, Vt, Yb);
  const int ptiles = (B_ * T_ / 16) * (EMB_ / 64);  // 8192 wave-tiles
  proj_kernel<<<ptiles / 8, 256, 0, stream>>>(Yb, Wub, bu, out);
}
